// HeteroTopologyEncoder_68642167324673
// MI455X (gfx1250) — matrix operand order repaired and checked by reading, not compiled, not run on hardware
//
#include <hip/hip_runtime.h>
#include <hip/hip_bf16.h>
#include <math.h>

// ---------------- problem constants (match reference) ----------------
static constexpr int NV_ = 30000, NR_ = 1000;
static constexpr int EVV_ = 400000, EVI_ = 120000, EIV_ = 120000;
static constexpr int VD_ = 16, RD_ = 8, ED_ = 2;
static constexpr int HID_ = 64, OUT_ = 64, H1_ = 2;
static constexpr int H1C_ = H1_ * HID_; // 128

typedef __attribute__((ext_vector_type(16))) _Float16 v16h;
typedef __attribute__((ext_vector_type(8)))  float    v8f;

// ---------------- small helpers ----------------
__device__ __forceinline__ float leaky02(float x) { return x > 0.f ? x : 0.2f * x; }

// order-preserving float <-> uint bijection for atomic float max
__device__ __forceinline__ unsigned fenc(float f) {
  unsigned u = __float_as_uint(f);
  return (u & 0x80000000u) ? ~u : (u | 0x80000000u);
}
__device__ __forceinline__ float fdec(unsigned u) {
  return (u & 0x80000000u) ? __uint_as_float(u ^ 0x80000000u) : __uint_as_float(~u);
}
static constexpr unsigned ENC_NEG_INF = 0x007FFFFFu; // fenc(-inf)

// ---------------- fill kernels ----------------
__global__ void fill_f32(float* p, long long n, float v) {
  long long i = (long long)blockIdx.x * blockDim.x + threadIdx.x;
  if (i < n) p[i] = v;
}
__global__ void fill_u32(unsigned* p, long long n, unsigned v) {
  long long i = (long long)blockIdx.x * blockDim.x + threadIdx.x;
  if (i < n) p[i] = v;
}

// ---------------- self-loop mean edge-attr (fill_value='mean') ----------------
__global__ void selfloop_accum(const int* __restrict__ dst, const float* __restrict__ ea,
                               int E, float* __restrict__ sumEa, float* __restrict__ deg) {
  int e = blockIdx.x * blockDim.x + threadIdx.x;
  if (e >= E) return;
  int d = dst[e];
  atomicAdd(&deg[d], 1.f);
  atomicAdd(&sumEa[2 * d + 0], ea[2 * e + 0]);
  atomicAdd(&sumEa[2 * d + 1], ea[2 * e + 1]);
}
__global__ void selfloop_mean(float* __restrict__ sumEa, const float* __restrict__ deg, int N) {
  int n = blockIdx.x * blockDim.x + threadIdx.x;
  if (n >= N) return;
  float dg = fmaxf(deg[n], 1.f);
  sumEa[2 * n + 0] /= dg;
  sumEa[2 * n + 1] /= dg;
}

// ---------------- WMMA GEMM: C[MxN] = A[MxK] @ B[KxN] ----------------
// fp32 in memory, f16 WMMA math, f32 accumulate. LDS-staged, K/N compile-time.
// Block = 256 threads = 8 waves; macro-tile 32(M) x 64(N); BK = 32.
template <int K, int N>
__global__ void __launch_bounds__(256)
wmma_gemm_f16(const float* __restrict__ A, const float* __restrict__ B,
              float* __restrict__ C, int M) {
  constexpr int BM = 32, BN = 64, BK = 32;
  constexpr int ASTR = BK + 8;  // LDS row stride (halves), padded
  constexpr int BSTR = BN + 8;
  __shared__ _Float16 As[BM * ASTR];
  __shared__ _Float16 Bs[BK * BSTR];

  constexpr int gridN = N / BN;
  int gm = blockIdx.x / gridN, gn = blockIdx.x % gridN;
  int t = threadIdx.x;
  int wave = t >> 5, lane = t & 31;
  int l16 = lane & 15, khalf = lane >> 4;
  int wm = wave >> 2, wn = wave & 3; // 2 x 4 wave grid of 16x16 tiles

  v8f acc = {};
  for (int kb = 0; kb < K; kb += BK) {
    // ---- global -> LDS, A tile (BM x BK), coalesced float4 loads ----
    {
      int r = t >> 3;              // 0..31
      int kq = (t & 7) << 2;       // 0,4,...,28
      int gr = gm * BM + r;
      int k = kb + kq;
      float4 v = make_float4(0.f, 0.f, 0.f, 0.f);
      if (gr < M && k < K)         // K % 4 == 0 -> whole chunk in range
        v = *(const float4*)(A + (size_t)gr * K + k);
      _Float16* p = &As[r * ASTR + kq];
      p[0] = (_Float16)v.x; p[1] = (_Float16)v.y;
      p[2] = (_Float16)v.z; p[3] = (_Float16)v.w;
    }
    // ---- global -> LDS, B tile (BK x BN), coalesced float4 loads ----
    {
      int k0 = t >> 4;             // 0..15
      int nq = (t & 15) << 2;      // 0,4,...,60
      int gc = gn * BN + nq;       // N % BN == 0 -> always in range
#pragma unroll
      for (int kk = 0; kk < 2; ++kk) {
        int k = k0 + kk * 16;
        float4 v = make_float4(0.f, 0.f, 0.f, 0.f);
        if (kb + k < K)
          v = *(const float4*)(B + (size_t)(kb + k) * N + gc);
        _Float16* p = &Bs[k * BSTR + nq];
        p[0] = (_Float16)v.x; p[1] = (_Float16)v.y;
        p[2] = (_Float16)v.z; p[3] = (_Float16)v.w;
      }
    }
    __syncthreads();
    // ---- fragments from LDS (ISA §7.12.2 layouts) ----
    v16h a, b;
    const _Float16* arow = &As[(wm * 16 + l16) * ASTR];
#pragma unroll
    for (int i = 0; i < 8; ++i) {
      a[i]     = arow[khalf * 8 + i];        // K = khalf*8 + i
      a[i + 8] = arow[16 + khalf * 8 + i];   // K = 16 + khalf*8 + i
    }
#pragma unroll
    for (int i = 0; i < 16; ++i)
      b[i] = Bs[(((i & 8) << 1) + khalf * 8 + (i & 7)) * BSTR + wn * 16 + l16];
    acc = __builtin_amdgcn_wmma_f32_16x16x32_f16(false, a, false, b, (short)0, acc,
                                                 false, false);
    __syncthreads();
  }
  // ---- store C (VGPR r holds M = r + 8*khalf, N = lane&15) ----
  int n = gn * BN + wn * 16 + l16;
  int mbase = gm * BM + wm * 16 + (khalf << 3);
  if (gm * BM + BM <= M) {
    // uniform full-tile fast path: no per-element predication
#pragma unroll
    for (int r = 0; r < 8; ++r)
      C[(size_t)(mbase + r) * N + n] = acc[r];
  } else {
#pragma unroll
    for (int r = 0; r < 8; ++r)
      if (mbase + r < M) C[(size_t)(mbase + r) * N + n] = acc[r];
  }
}

// ---------------- edge pass 1: logits + per-dst max (wave per edge) ----------------
__global__ void edge_logits(const int* __restrict__ src, const int* __restrict__ dst,
                            const float* __restrict__ ea, const float* __restrict__ meanEa,
                            int E, int nself,
                            const float* __restrict__ hl, const float* __restrict__ hr,
                            const float* __restrict__ We, const float* __restrict__ att,
                            int H, int C, float* __restrict__ logits,
                            unsigned* __restrict__ mEnc) {
  int wavesPerBlock = blockDim.x >> 5;
  int e = blockIdx.x * wavesPerBlock + (threadIdx.x >> 5);
  int lane = threadIdx.x & 31;
  int Etot = E + nself;
  if (e >= Etot) return;
  int s, d; float e0, e1;
  if (e < E) { s = src[e]; d = dst[e]; e0 = ea[2 * e]; e1 = ea[2 * e + 1]; }
  else       { s = e - E;  d = s;      e0 = meanEa[2 * s]; e1 = meanEa[2 * s + 1]; }
  int HC = H * C;
  const float* hls = hl + (size_t)s * HC;
  const float* hrd = hr + (size_t)d * HC;
  for (int h = 0; h < H; ++h) {
    float part = 0.f;
    for (int c = lane; c < C; c += 32) {
      int i = h * C + c;
      float g = hls[i] + hrd[i] + e0 * We[i] + e1 * We[HC + i];
      part += att[h * C + c] * leaky02(g);
    }
#pragma unroll
    for (int off = 16; off > 0; off >>= 1) part += __shfl_xor(part, off, 32);
    if (lane == 0) {
      logits[(size_t)e * H + h] = part;
      atomicMax(&mEnc[(size_t)d * H + h], fenc(part));
    }
  }
}

// ---------------- edge pass 2: ex = exp(logit - m[dst]); den[dst] += ex ----------------
__global__ void edge_exp(const int* __restrict__ dst, int E, int nself, int H,
                         const unsigned* __restrict__ mEnc, float* __restrict__ logits,
                         float* __restrict__ den) {
  long long idx = (long long)blockIdx.x * blockDim.x + threadIdx.x;
  long long total = (long long)(E + nself) * H;
  if (idx >= total) return;
  int e = (int)(idx / H), h = (int)(idx % H);
  int d = (e < E) ? dst[e] : (e - E);
  float m = fdec(mEnc[(size_t)d * H + h]);
  if (!(m >= -3.0e38f && m <= 3.0e38f)) m = 0.f;  // mirror where(isfinite(m), m, 0)
  float ex = expf(logits[idx] - m);
  logits[idx] = ex;
  atomicAdd(&den[(size_t)d * H + h], ex);
}

// ---------------- edge pass 3: out[dst] += alpha * hl[src] ----------------
__global__ void edge_scatter(const int* __restrict__ src, const int* __restrict__ dst,
                             int E, int nself, int H, int C,
                             const float* __restrict__ hl, const float* __restrict__ ex,
                             const float* __restrict__ den, float* __restrict__ out) {
  long long idx = (long long)blockIdx.x * blockDim.x + threadIdx.x;
  int HC = H * C;
  long long total = (long long)(E + nself) * HC;
  if (idx >= total) return;
  int e = (int)(idx / HC);
  int i = (int)(idx % HC);
  int h = i / C;
  int s, d;
  if (e < E) { s = src[e]; d = dst[e]; } else { s = e - E; d = s; }
  float alpha = ex[(size_t)e * H + h] / (den[(size_t)d * H + h] + 1e-16f);
  atomicAdd(&out[(size_t)d * HC + i], alpha * hl[(size_t)s * HC + i]);
}

// ---------------- bias (+ optional second bias) and optional ELU ----------------
__global__ void bias_act(float* __restrict__ x, long long total, int HC,
                         const float* __restrict__ b1, const float* __restrict__ b2,
                         int doElu) {
  long long idx = (long long)blockIdx.x * blockDim.x + threadIdx.x;
  if (idx >= total) return;
  int i = (int)(idx % HC);
  float v = x[idx] + b1[i] + (b2 ? b2[i] : 0.f);
  if (doElu) v = v > 0.f ? v : (expf(v) - 1.f);
  x[idx] = v;
}

// ---------------- LayerNorm over 64 channels (wave per node) ----------------
__global__ void layernorm64(const float* __restrict__ x, const float* __restrict__ g,
                            const float* __restrict__ b, float* __restrict__ out, int N) {
  int wavesPerBlock = blockDim.x >> 5;
  int n = blockIdx.x * wavesPerBlock + (threadIdx.x >> 5);
  int lane = threadIdx.x & 31;
  if (n >= N) return;
  float v0 = x[(size_t)n * 64 + lane];
  float v1 = x[(size_t)n * 64 + 32 + lane];
  float s = v0 + v1;
#pragma unroll
  for (int off = 16; off > 0; off >>= 1) s += __shfl_xor(s, off, 32);
  float mean = s * (1.f / 64.f);
  float d0 = v0 - mean, d1 = v1 - mean;
  float vs = d0 * d0 + d1 * d1;
#pragma unroll
  for (int off = 16; off > 0; off >>= 1) vs += __shfl_xor(vs, off, 32);
  float rstd = rsqrtf(vs * (1.f / 64.f) + 1e-5f);
  out[(size_t)n * 64 + lane]      = d0 * rstd * g[lane] + b[lane];
  out[(size_t)n * 64 + 32 + lane] = d1 * rstd * g[32 + lane] + b[32 + lane];
}

// ---------------- host-side helpers ----------------
template <int K, int N>
static inline void launch_gemm_t(hipStream_t s, const float* A, const float* B, float* C,
                                 int M) {
  int blocks = ((M + 31) / 32) * (N / 64);
  wmma_gemm_f16<K, N><<<blocks, 256, 0, s>>>(A, B, C, M);
}

static void launch_gemm(hipStream_t s, const float* A, const float* B, float* C,
                        int M, int K, int N) {
  if (K == 16 && N == 128)      launch_gemm_t<16, 128>(s, A, B, C, M);
  else if (K == 8 && N == 128)  launch_gemm_t<8, 128>(s, A, B, C, M);
  else if (K == 128 && N == 64) launch_gemm_t<128, 64>(s, A, B, C, M);
}

static void run_relation(hipStream_t s,
                         const float* xs, int Ns, int Ks,
                         const float* xd, int Nd, int Kd,
                         const int* src, const int* dst, const float* ea,
                         int E, int nself, const float* meanEa,
                         const float* Wl, const float* Wr, const float* We,
                         const float* att, int H, int C,
                         float* hl, float* hr, float* exb, unsigned* mEnc, float* den,
                         float* out) {
  int HC = H * C;
  launch_gemm(s, xs, Wl, hl, Ns, Ks, HC);
  launch_gemm(s, xd, Wr, hr, Nd, Kd, HC);
  long long nm = (long long)Nd * H;
  fill_u32<<<(unsigned)((nm + 255) / 256), 256, 0, s>>>(mEnc, nm, ENC_NEG_INF);
  fill_f32<<<(unsigned)((nm + 255) / 256), 256, 0, s>>>(den, nm, 0.f);
  int Etot = E + nself;
  edge_logits<<<(Etot + 7) / 8, 256, 0, s>>>(src, dst, ea, meanEa, E, nself,
                                             hl, hr, We, att, H, C, exb, mEnc);
  long long ne = (long long)Etot * H;
  edge_exp<<<(unsigned)((ne + 255) / 256), 256, 0, s>>>(dst, E, nself, H, mEnc, exb, den);
  long long nsc = (long long)Etot * HC;
  edge_scatter<<<(unsigned)((nsc + 255) / 256), 256, 0, s>>>(src, dst, E, nself, H, C,
                                                             hl, exb, den, out);
}

extern "C" void kernel_launch(void* const* d_in, const int* in_sizes, int n_in,
                              void* d_out, int out_size, void* d_ws, size_t ws_size,
                              hipStream_t stream) {
  (void)in_sizes; (void)n_in; (void)out_size; (void)ws_size;
  // ---- inputs (setup_inputs order) ----
  const float* x_veh  = (const float*)d_in[0];
  const float* x_rsu  = (const float*)d_in[1];
  const float* ea_vv  = (const float*)d_in[2];
  const float* ea_vi  = (const float*)d_in[3];
  const float* ea_iv  = (const float*)d_in[4];
  const int*   s_vv   = (const int*)d_in[5];
  const int*   d_vv   = (const int*)d_in[6];
  const int*   s_vi   = (const int*)d_in[7];
  const int*   d_vi   = (const int*)d_in[8];
  const int*   s_iv   = (const int*)d_in[9];
  const int*   d_iv   = (const int*)d_in[10];
  const float* Wl_vv1 = (const float*)d_in[11]; const float* Wr_vv1 = (const float*)d_in[12];
  const float* We_vv1 = (const float*)d_in[13]; const float* att_vv1 = (const float*)d_in[14];
  const float* b_vv1  = (const float*)d_in[15];
  const float* Wl_vi1 = (const float*)d_in[16]; const float* Wr_vi1 = (const float*)d_in[17];
  const float* We_vi1 = (const float*)d_in[18]; const float* att_vi1 = (const float*)d_in[19];
  const float* b_vi1  = (const float*)d_in[20];
  const float* Wl_iv1 = (const float*)d_in[21]; const float* Wr_iv1 = (const float*)d_in[22];
  const float* We_iv1 = (const float*)d_in[23]; const float* att_iv1 = (const float*)d_in[24];
  const float* b_iv1  = (const float*)d_in[25];
  const float* Wl_vv2 = (const float*)d_in[26]; const float* Wr_vv2 = (const float*)d_in[27];
  const float* We_vv2 = (const float*)d_in[28]; const float* att_vv2 = (const float*)d_in[29];
  const float* b_vv2  = (const float*)d_in[30];
  const float* Wl_vi2 = (const float*)d_in[31]; const float* Wr_vi2 = (const float*)d_in[32];
  const float* We_vi2 = (const float*)d_in[33]; const float* att_vi2 = (const float*)d_in[34];
  const float* b_vi2  = (const float*)d_in[35];
  const float* Wl_iv2 = (const float*)d_in[36]; const float* Wr_iv2 = (const float*)d_in[37];
  const float* We_iv2 = (const float*)d_in[38]; const float* att_iv2 = (const float*)d_in[39];
  const float* b_iv2  = (const float*)d_in[40];
  const float* ln_vg = (const float*)d_in[41]; const float* ln_vb = (const float*)d_in[42];
  const float* ln_rg = (const float*)d_in[43]; const float* ln_rb = (const float*)d_in[44];
  float* out = (float*)d_out;

  // ---- workspace layout (floats) ----
  float* ws = (float*)d_ws;
  size_t off = 0;
  auto take = [&](size_t n) { float* p = ws + off; off += n; return p; };
  float*    v1     = take((size_t)NV_ * H1C_);
  float*    r1     = take((size_t)NR_ * H1C_);
  float*    v2     = take((size_t)NV_ * OUT_);
  float*    r2     = take((size_t)NR_ * OUT_);
  float*    hl     = take((size_t)NV_ * H1C_);
  float*    hr     = take((size_t)NV_ * H1C_);
  float*    exb    = take((size_t)(EVV_ + NV_) * H1_);
  unsigned* mEnc   = (unsigned*)take((size_t)NV_ * H1_);
  float*    den    = take((size_t)NV_ * H1_);
  float*    meanEa = take((size_t)NV_ * ED_);
  float*    deg    = take((size_t)NV_);

  // ---- self-loop mean edge attrs for v2v (shared by both layers) ----
  {
    long long n2 = (long long)NV_ * ED_;
    fill_f32<<<(unsigned)((n2 + 255) / 256), 256, 0, stream>>>(meanEa, n2, 0.f);
    fill_f32<<<(NV_ + 255) / 256, 256, 0, stream>>>(deg, NV_, 0.f);
    selfloop_accum<<<(EVV_ + 255) / 256, 256, 0, stream>>>(d_vv, ea_vv, EVV_, meanEa, deg);
    selfloop_mean<<<(NV_ + 255) / 256, 256, 0, stream>>>(meanEa, deg, NV_);
  }

  // ---- zero accumulators ----
  {
    long long nv1 = (long long)NV_ * H1C_, nr1 = (long long)NR_ * H1C_;
    long long nv2 = (long long)NV_ * OUT_, nr2 = (long long)NR_ * OUT_;
    fill_f32<<<(unsigned)((nv1 + 255) / 256), 256, 0, stream>>>(v1, nv1, 0.f);
    fill_f32<<<(unsigned)((nr1 + 255) / 256), 256, 0, stream>>>(r1, nr1, 0.f);
    fill_f32<<<(unsigned)((nv2 + 255) / 256), 256, 0, stream>>>(v2, nv2, 0.f);
    fill_f32<<<(unsigned)((nr2 + 255) / 256), 256, 0, stream>>>(r2, nr2, 0.f);
  }

  // ================= layer 1 (H=2, C=64, concat) =================
  run_relation(stream, x_veh, NV_, VD_, x_veh, NV_, VD_,
               s_vv, d_vv, ea_vv, EVV_, NV_, meanEa,
               Wl_vv1, Wr_vv1, We_vv1, att_vv1, H1_, HID_,
               hl, hr, exb, mEnc, den, v1);
  run_relation(stream, x_rsu, NR_, RD_, x_veh, NV_, VD_,
               s_iv, d_iv, ea_iv, EIV_, 0, meanEa,
               Wl_iv1, Wr_iv1, We_iv1, att_iv1, H1_, HID_,
               hl, hr, exb, mEnc, den, v1);
  run_relation(stream, x_veh, NV_, VD_, x_rsu, NR_, RD_,
               s_vi, d_vi, ea_vi, EVI_, 0, meanEa,
               Wl_vi1, Wr_vi1, We_vi1, att_vi1, H1_, HID_,
               hl, hr, exb, mEnc, den, r1);
  {
    long long nv1 = (long long)NV_ * H1C_, nr1 = (long long)NR_ * H1C_;
    bias_act<<<(unsigned)((nv1 + 255) / 256), 256, 0, stream>>>(v1, nv1, H1C_, b_vv1, b_iv1, 1);
    bias_act<<<(unsigned)((nr1 + 255) / 256), 256, 0, stream>>>(r1, nr1, H1C_, b_vi1, nullptr, 1);
  }

  // ================= layer 2 (H=1, C=64) =================
  run_relation(stream, v1, NV_, H1C_, v1, NV_, H1C_,
               s_vv, d_vv, ea_vv, EVV_, NV_, meanEa,
               Wl_vv2, Wr_vv2, We_vv2, att_vv2, 1, OUT_,
               hl, hr, exb, mEnc, den, v2);
  run_relation(stream, r1, NR_, H1C_, v1, NV_, H1C_,
               s_iv, d_iv, ea_iv, EIV_, 0, meanEa,
               Wl_iv2, Wr_iv2, We_iv2, att_iv2, 1, OUT_,
               hl, hr, exb, mEnc, den, v2);
  run_relation(stream, v1, NV_, H1C_, r1, NR_, H1C_,
               s_vi, d_vi, ea_vi, EVI_, 0, meanEa,
               Wl_vi2, Wr_vi2, We_vi2, att_vi2, 1, OUT_,
               hl, hr, exb, mEnc, den, r2);
  {
    long long nv2 = (long long)NV_ * OUT_, nr2 = (long long)NR_ * OUT_;
    bias_act<<<(unsigned)((nv2 + 255) / 256), 256, 0, stream>>>(v2, nv2, OUT_, b_vv2, b_iv2, 0);
    bias_act<<<(unsigned)((nr2 + 255) / 256), 256, 0, stream>>>(r2, nr2, OUT_, b_vi2, nullptr, 0);
  }

  // ---- LayerNorm + concat into d_out ----
  layernorm64<<<(NV_ + 7) / 8, 256, 0, stream>>>(v2, ln_vg, ln_vb, out, NV_);
  layernorm64<<<(NR_ + 7) / 8, 256, 0, stream>>>(r2, ln_rg, ln_rb, out + (size_t)NV_ * OUT_, NR_);
}
